// MultiHeadSelfAttention_11012296147280
// MI455X (gfx1250) — matrix operand order repaired and checked by reading, not compiled, not run on hardware
//
#include <hip/hip_runtime.h>

#define Bsz  4
#define Tseq 2048
#define Dm   512
#define Hh   8
#define HD   64

typedef __attribute__((ext_vector_type(16))) __bf16 v16bf;
typedef __attribute__((ext_vector_type(8)))  float  v8f;
typedef unsigned short u16;
typedef unsigned int   u32;

__device__ __forceinline__ u16 f2bf(float f) {
  union { float f; u32 u; } v; v.f = f;
  u32 r = v.u + 0x7FFFu + ((v.u >> 16) & 1u);   // round-to-nearest-even
  return (u16)(r >> 16);
}

union FragU { uint4 q[2]; v16bf v; };

__device__ __forceinline__ v16bf load_frag_split(const u16* lo, const u16* hi) {
  FragU u;
  u.q[0] = *(const uint4*)lo;
  u.q[1] = *(const uint4*)hi;
  return u.v;
}
__device__ __forceinline__ v16bf load_frag16(const u16* p) {
  return load_frag_split(p, p + 8);
}

// A-frag (16x32 bf16, M x K), src row-major with leading dim ld, tile rows at `base`.
// Layout: lanes 0-15 (row M=lane): K0..15;
//         lanes 16-31 (row M=lane-16): K16..31 (matches B-frag order).
__device__ __forceinline__ v16bf load_afrag(const u16* base, int ld, int k0, int lane) {
  int m = lane & 15, half = lane >> 4;
  const u16* row = base + (size_t)m * ld;
  return load_frag_split(row + k0 + 16 * half, row + k0 + 16 * half + 8);
}
// B-frag (32x16 bf16, K x N), src stored column-contiguous: B^T row-major [N][ld].
// Layout used here (same k order as A-frag): lane = column N; lanes 0-15 hold K=0..15, lanes 16-31 hold K=16..31.
__device__ __forceinline__ v16bf load_bfrag(const u16* baseT, int ld, int k0, int lane) {
  int n = lane & 15, half = lane >> 4;
  return load_frag16(baseT + (size_t)n * ld + k0 + 16 * half);
}

__device__ __forceinline__ v8f wmma_bf16(v16bf a, v16bf b, v8f c) {
  return __builtin_amdgcn_wmma_f32_16x16x32_bf16(false, a, false, b, (short)0, c, false, false);
}

// ---------------------------------------------------------------- converts
__global__ void k_cvt_bf16(const float* __restrict__ in, u16* __restrict__ out, int n) {
  int i = blockIdx.x * blockDim.x + threadIdx.x;
  if (i < n) out[i] = f2bf(in[i]);
}

// in[K][N] f32 (row-major) -> out[N][K] bf16  (so B columns become contiguous)
__global__ void k_transpose_bf16(const float* __restrict__ in, u16* __restrict__ out,
                                 int K, int N) {
  int i = blockIdx.x * blockDim.x + threadIdx.x;
  if (i < K * N) {
    int n = i / K, k = i % K;
    out[(size_t)n * K + k] = f2bf(in[(size_t)k * N + n]);
  }
}

// ---------------------------------------------------------------- QKV GEMM
// xb [B*T, D] bf16  @  Wqkv^T [3D, D] bf16 -> Q,K [b,h,t,d] bf16 ; V^T [b,h,d,t] bf16
__global__ void k_qkv_gemm(const u16* __restrict__ xb, const u16* __restrict__ wT,
                           u16* __restrict__ Qb, u16* __restrict__ Kb, u16* __restrict__ Vt) {
  int lane = threadIdx.x;
  int m0 = blockIdx.x * 16;
  int n0 = blockIdx.y * 64;
  v8f acc[4] = {};
  for (int k0 = 0; k0 < Dm; k0 += 32) {
    v16bf a = load_afrag(xb + (size_t)m0 * Dm, Dm, k0, lane);
#pragma unroll
    for (int t = 0; t < 4; t++) {
      v16bf b = load_bfrag(wT + (size_t)(n0 + t * 16) * Dm, Dm, k0, lane);
      acc[t] = wmma_bf16(a, b, acc[t]);
    }
  }
  int half = lane >> 4, nl = lane & 15;
#pragma unroll
  for (int t = 0; t < 4; t++) {
    int n   = n0 + t * 16 + nl;
    int sec = n >> 9;           // 0=Q 1=K 2=V
    int nn  = n & 511;
    int h = nn >> 6, d = nn & 63;
#pragma unroll
    for (int r = 0; r < 8; r++) {
      int m  = m0 + r + 8 * half;                  // D-layout row
      int b  = m >> 11, tt = m & (Tseq - 1);
      int bh = b * Hh + h;
      u16 val = f2bf(acc[t][r]);
      if (sec == 0)      Qb[((size_t)bh * Tseq + tt) * HD + d] = val;
      else if (sec == 1) Kb[((size_t)bh * Tseq + tt) * HD + d] = val;
      else               Vt[((size_t)bh * HD + d) * Tseq + tt] = val;  // transposed
    }
  }
}

// ---------------------------------------------------------------- flash attention
// one wave per (b,h, 16-query tile); streams 32 keys/iteration
__global__ void k_attn(const u16* __restrict__ Qb, const u16* __restrict__ Kb,
                       const u16* __restrict__ Vt, u16* __restrict__ Yb) {
  __shared__ float Sld[16 * 32];
  __shared__ u16   Pld[16 * 32];
  __shared__ float scaleLd[16];
  __shared__ float sumLd[16];

  int lane = threadIdx.x;
  int nl = lane & 15, half = lane >> 4;
  int qtiles = Tseq / 16;
  int bh = blockIdx.x / qtiles;
  int q0 = (blockIdx.x % qtiles) * 16;
  int b = bh / Hh, h = bh % Hh;

  const u16* Qh = Qb + (size_t)bh * Tseq * HD;
  const u16* Kh = Kb + (size_t)bh * Tseq * HD;
  const u16* Vh = Vt + (size_t)bh * HD * Tseq;

  v16bf qa0 = load_afrag(Qh + (size_t)q0 * HD, HD, 0,  lane);
  v16bf qa1 = load_afrag(Qh + (size_t)q0 * HD, HD, 32, lane);

  v8f   O[4] = {};
  float run_max = -3.0e38f, run_sum = 0.0f;

  int nblocks = (q0 + 16 + 31) / 32;            // causal: skip fully-masked key blocks
  for (int jb = 0; jb < nblocks; jb++) {
    int j = jb * 32;
    v8f s0 = {}, s1 = {};
    s0 = wmma_bf16(qa0, load_bfrag(Kh + (size_t)j * HD,        HD, 0,  lane), s0);
    s0 = wmma_bf16(qa1, load_bfrag(Kh + (size_t)j * HD,        HD, 32, lane), s0);
    s1 = wmma_bf16(qa0, load_bfrag(Kh + (size_t)(j + 16) * HD, HD, 0,  lane), s1);
    s1 = wmma_bf16(qa1, load_bfrag(Kh + (size_t)(j + 16) * HD, HD, 32, lane), s1);

    // scale + causal mask in D-layout regs, spill to LDS (layout change for P@V)
#pragma unroll
    for (int r = 0; r < 8; r++) {
      int m = r + 8 * half;
      int q = q0 + m;
      float v0 = (j + nl      <= q) ? s0[r] * 0.125f : -3.0e38f;
      float v1 = (j + 16 + nl <= q) ? s1[r] * 0.125f : -3.0e38f;
      Sld[m * 32 + nl]      = v0;
      Sld[m * 32 + 16 + nl] = v1;
    }
    __syncthreads();

    // online softmax: lanes 0-15 each own one query row
    if (lane < 16) {
      int m = lane;
      float sv[32], tmax = -3.0e38f;
#pragma unroll
      for (int c = 0; c < 32; c++) { sv[c] = Sld[m * 32 + c]; tmax = fmaxf(tmax, sv[c]); }
      float nmax   = fmaxf(run_max, tmax);
      float factor = __expf(run_max - nmax);
      float psum   = 0.0f;
#pragma unroll
      for (int c = 0; c < 32; c++) {
        float p = __expf(sv[c] - nmax);
        psum += p;
        Pld[m * 32 + c] = f2bf(p);
      }
      run_sum = run_sum * factor + psum;
      run_max = nmax;
      scaleLd[m] = factor;
    }
    __syncthreads();

    // rescale accumulator by per-row factor
    float fac[8];
#pragma unroll
    for (int r = 0; r < 8; r++) fac[r] = scaleLd[r + 8 * half];
#pragma unroll
    for (int t = 0; t < 4; t++)
#pragma unroll
      for (int r = 0; r < 8; r++) O[t][r] *= fac[r];

    // O += P (16x32) @ V (32x64): V^T rows are contiguous columns of B
    v16bf pa = load_afrag(Pld, 32, 0, lane);
#pragma unroll
    for (int t = 0; t < 4; t++) {
      v16bf vb = load_frag16(Vh + (size_t)(t * 16 + nl) * Tseq + j + 16 * half);
      O[t] = wmma_bf16(pa, vb, O[t]);
    }
    __syncthreads();
  }

  if (lane < 16) sumLd[lane] = run_sum;
  __syncthreads();

#pragma unroll
  for (int r = 0; r < 8; r++) {
    int m = r + 8 * half;
    float inv = 1.0f / sumLd[m];
    int q = q0 + m;
#pragma unroll
    for (int t = 0; t < 4; t++)
      Yb[((size_t)(b * Tseq + q) * Dm) + h * HD + t * 16 + nl] = f2bf(O[t][r] * inv);
  }
}

// ---------------------------------------------------------------- output projection
__global__ void k_out_gemm(const u16* __restrict__ yb, const u16* __restrict__ wT,
                           float* __restrict__ out) {
  int lane = threadIdx.x;
  int m0 = blockIdx.x * 16;
  int n0 = blockIdx.y * 64;
  v8f acc[4] = {};
  for (int k0 = 0; k0 < Dm; k0 += 32) {
    v16bf a = load_afrag(yb + (size_t)m0 * Dm, Dm, k0, lane);
#pragma unroll
    for (int t = 0; t < 4; t++) {
      v16bf b = load_bfrag(wT + (size_t)(n0 + t * 16) * Dm, Dm, k0, lane);
      acc[t] = wmma_bf16(a, b, acc[t]);
    }
  }
  int half = lane >> 4, nl = lane & 15;
#pragma unroll
  for (int t = 0; t < 4; t++)
#pragma unroll
    for (int r = 0; r < 8; r++) {
      int m = m0 + r + 8 * half;
      out[(size_t)m * Dm + n0 + t * 16 + nl] = acc[t][r];
    }
}

// ---------------------------------------------------------------- launch
extern "C" void kernel_launch(void* const* d_in, const int* in_sizes, int n_in,
                              void* d_out, int out_size, void* d_ws, size_t ws_size,
                              hipStream_t stream) {
  const float* x    = (const float*)d_in[0];
  // d_in[1] = attn_mask (causal tril) — applied analytically in k_attn
  const float* Wqkv = (const float*)d_in[2];
  const float* Wout = (const float*)d_in[3];
  float* out = (float*)d_out;

  char* ws = (char*)d_ws;
  u16* xb    = (u16*)(ws);                         //  8 MiB  x bf16 [B*T, D]
  u16* wqkvT = (u16*)(ws + 8388608);               //  1.5 MiB Wqkv^T [3D, D]
  u16* woutT = (u16*)(ws + 9961472);               //  0.5 MiB Wout^T [D, D]
  u16* Qb    = (u16*)(ws + 10485760);              //  8 MiB  Q [b,h,t,d]
  u16* Kb    = (u16*)(ws + 18874368);              //  8 MiB  K [b,h,t,d]
  u16* Vt    = (u16*)(ws + 27262976);              //  8 MiB  V^T [b,h,d,t]
  u16* Yb    = (u16*)(ws + 35651584);              //  8 MiB  attn out bf16 [B*T, D]

  int nx = Bsz * Tseq * Dm;
  k_cvt_bf16<<<(nx + 255) / 256, 256, 0, stream>>>(x, xb, nx);
  k_transpose_bf16<<<(Dm * 3 * Dm + 255) / 256, 256, 0, stream>>>(Wqkv, wqkvT, Dm, 3 * Dm);
  k_transpose_bf16<<<(Dm * Dm + 255) / 256, 256, 0, stream>>>(Wout, woutT, Dm, Dm);

  dim3 g1(Bsz * Tseq / 16, (3 * Dm) / 64);
  k_qkv_gemm<<<g1, 32, 0, stream>>>(xb, wqkvT, Qb, Kb, Vt);

  k_attn<<<Bsz * Hh * (Tseq / 16), 32, 0, stream>>>(Qb, Kb, Vt, Yb);

  dim3 g2(Bsz * Tseq / 16, Dm / 64);
  k_out_gemm<<<g2, 32, 0, stream>>>(Yb, woutT, out);
}
